// BiMambaBlock_28741921144930
// MI455X (gfx1250) — matrix operand order repaired and checked by reading, not compiled, not run on hardware
//
#include <hip/hip_runtime.h>
#include <hip/hip_bf16.h>
#include <stdint.h>

// ---------------------------------------------------------------------------
// Types for CDNA5 WMMA (wave32, 16x16x32 bf16 -> f32)
// ---------------------------------------------------------------------------
typedef __attribute__((ext_vector_type(16))) __bf16 v16bf;
typedef __attribute__((ext_vector_type(8)))  float  v8f;

union FragBF16 { uint32_t u[8]; v16bf v; };

// ---------------------------------------------------------------------------
// Generic f32 -> bf16 conversion (weights + activations)
// ---------------------------------------------------------------------------
__global__ void f32_to_bf16_kernel(const float* __restrict__ src,
                                   __hip_bfloat16* __restrict__ dst, size_t n) {
  size_t i = (size_t)blockIdx.x * blockDim.x + threadIdx.x;
  if (i < n) dst[i] = __float2bfloat16(src[i]);
}

// f32 -> bf16 with zero padding of rows >= rows_src (for N padded to mult of 64)
__global__ void f32_to_bf16_pad_kernel(const float* __restrict__ src,
                                       __hip_bfloat16* __restrict__ dst,
                                       int rows_src, int cols, size_t n) {
  size_t i = (size_t)blockIdx.x * blockDim.x + threadIdx.x;
  if (i >= n) return;
  size_t r = i / cols;
  size_t c = i % cols;
  float v = (r < (size_t)rows_src) ? src[r * cols + c] : 0.f;
  dst[i] = __float2bfloat16(v);
}

// ---------------------------------------------------------------------------
// LayerNorm: one block per row of d_model=1024; writes fp32 + bf16 copies
// ---------------------------------------------------------------------------
__global__ __launch_bounds__(256) void layernorm_kernel(
    const float* __restrict__ x, const float* __restrict__ gam,
    const float* __restrict__ bet, float* __restrict__ xn,
    __hip_bfloat16* __restrict__ xn_bf, int dm) {
  int row = blockIdx.x;
  int tid = threadIdx.x;
  const float* xr = x + (size_t)row * dm;
  __shared__ float s1[256], s2[256];
  float a0 = 0.f, a1 = 0.f;
  for (int i = tid; i < dm; i += 256) { float v = xr[i]; a0 += v; a1 += v * v; }
  s1[tid] = a0; s2[tid] = a1;
  __syncthreads();
  for (int off = 128; off > 0; off >>= 1) {
    if (tid < off) { s1[tid] += s1[tid + off]; s2[tid] += s2[tid + off]; }
    __syncthreads();
  }
  float mean = s1[0] / dm;
  float var  = s2[0] / dm - mean * mean;
  float inv  = rsqrtf(var + 1e-5f);
  for (int i = tid; i < dm; i += 256) {
    float v = (xr[i] - mean) * inv * gam[i] + bet[i];
    xn[(size_t)row * dm + i]    = v;
    xn_bf[(size_t)row * dm + i] = __float2bfloat16(v);
  }
}

// ---------------------------------------------------------------------------
// bf16 GEMM:  C[M,N] = A[M,K] @ W[N,K]^T  (fp32 accumulate, v_wmma 16x16x32)
// One wave per 16(M) x 64(N) strip: A fragment reused across 4 WMMAs.
// STRICT requirements: M%16==0, N%64==0, K%32==0.
// Subtile count is a compile-time constant 4 so acc[] stays in fixed VGPRs
// (no v_movrel indexed-register traffic around the WMMAs).
// ---------------------------------------------------------------------------
__global__ __launch_bounds__(256) void gemm_bf16_nt_kernel(
    const __hip_bfloat16* __restrict__ A, const __hip_bfloat16* __restrict__ W,
    float* __restrict__ C, int M, int N, int K) {
  int tid  = threadIdx.x;
  int wave = tid >> 5;
  int lane = tid & 31;
  int g    = lane >> 4;   // lane half (K-group selector)
  int mr   = lane & 15;   // row (A) / col (B,C) within tile

  int tm  = blockIdx.x;                    // M tile index
  int tnb = (blockIdx.y * 8 + wave) * 4;   // base N tile (units of 16 cols)
  if (tnb * 16 >= N) return;               // N%64==0 -> wave fully in or out

  const __hip_bfloat16* Arow = A + (size_t)(tm * 16 + mr) * K;
  const __hip_bfloat16* Wrow0 = W + (size_t)((tnb + 0) * 16 + mr) * K + 16 * g;
  const __hip_bfloat16* Wrow1 = W + (size_t)((tnb + 1) * 16 + mr) * K + 16 * g;
  const __hip_bfloat16* Wrow2 = W + (size_t)((tnb + 2) * 16 + mr) * K + 16 * g;
  const __hip_bfloat16* Wrow3 = W + (size_t)((tnb + 3) * 16 + mr) * K + 16 * g;

  v8f acc0 = {}, acc1 = {}, acc2 = {}, acc3 = {};

  for (int k0 = 0; k0 < K; k0 += 32) {
    __builtin_prefetch(Arow + k0 + 64, 0, 0);   // global_prefetch_b8
    // A fragment: 16x32 bf16; lane group g holds K = 16g + {0..15}
    FragBF16 ua;
#pragma unroll
    for (int j = 0; j < 8; j++)
      ua.u[j] = *(const uint32_t*)(Arow + k0 + 16 * g + 2 * j);

    // B fragments: B[k][n] = W[n][k]; lane holds col n, K = 16g + {0..15}
    FragBF16 ub0, ub1, ub2, ub3;
#pragma unroll
    for (int j = 0; j < 8; j++) {
      ub0.u[j] = *(const uint32_t*)(Wrow0 + k0 + 2 * j);
      ub1.u[j] = *(const uint32_t*)(Wrow1 + k0 + 2 * j);
      ub2.u[j] = *(const uint32_t*)(Wrow2 + k0 + 2 * j);
      ub3.u[j] = *(const uint32_t*)(Wrow3 + k0 + 2 * j);
    }

    acc0 = __builtin_amdgcn_wmma_f32_16x16x32_bf16(false, ua.v, false, ub0.v,
                                                   (short)0, acc0, false, false);
    acc1 = __builtin_amdgcn_wmma_f32_16x16x32_bf16(false, ua.v, false, ub1.v,
                                                   (short)0, acc1, false, false);
    acc2 = __builtin_amdgcn_wmma_f32_16x16x32_bf16(false, ua.v, false, ub2.v,
                                                   (short)0, acc2, false, false);
    acc3 = __builtin_amdgcn_wmma_f32_16x16x32_bf16(false, ua.v, false, ub3.v,
                                                   (short)0, acc3, false, false);
  }

  // C layout: VGPR i -> row (i + 8*g), col = mr (within 16x16 tile)
  size_t rowBase = (size_t)(tm * 16 + 8 * g) * N;
#pragma unroll
  for (int i = 0; i < 8; i++) {
    size_t r = rowBase + (size_t)i * N;
    C[r + (tnb + 0) * 16 + mr] = acc0[i];
    C[r + (tnb + 1) * 16 + mr] = acc1[i];
    C[r + (tnb + 2) * 16 + mr] = acc2[i];
    C[r + (tnb + 3) * 16 + mr] = acc3[i];
  }
}

// ---------------------------------------------------------------------------
// Depthwise causal conv (d_conv=4) + SiLU. rev=1 flips the time direction
// (backward Mamba) without materializing a reversed sequence.
// u lives in cols [0, DI) of uz (row stride 2*DI). Writes fp32 + bf16.
// ---------------------------------------------------------------------------
__global__ void conv_silu_kernel(const float* __restrict__ uz,
                                 const float* __restrict__ cw,
                                 const float* __restrict__ cb,
                                 float* __restrict__ uc,
                                 __hip_bfloat16* __restrict__ uc_bf,
                                 int L, int DI, int rev, size_t n) {
  size_t idx = (size_t)blockIdx.x * blockDim.x + threadIdx.x;
  if (idx >= n) return;
  int d = (int)(idx % DI);
  size_t bt = idx / DI;
  int t = (int)(bt % L);
  int b = (int)(bt / L);
  float acc = cb[d];
#pragma unroll
  for (int k = 0; k < 4; k++) {
    int ts = rev ? (t + 3 - k) : (t - 3 + k);
    if (ts >= 0 && ts < L)
      acc += cw[d * 4 + k] * uz[((size_t)b * L + ts) * (size_t)(2 * DI) + d];
  }
  float s = acc / (1.f + expf(-acc));   // SiLU
  uc[idx]    = s;
  uc_bf[idx] = __float2bfloat16(s);
}

// ---------------------------------------------------------------------------
// Extract first dt_rank=64 cols of dbc (ld 128) into a packed bf16 matrix
// ---------------------------------------------------------------------------
__global__ void extract_dtin_kernel(const float* __restrict__ dbc,
                                    __hip_bfloat16* __restrict__ dtin,
                                    size_t n) {
  size_t idx = (size_t)blockIdx.x * blockDim.x + threadIdx.x;
  if (idx >= n) return;
  size_t r = idx >> 6;      // /64
  int c = (int)(idx & 63);
  dtin[idx] = __float2bfloat16(dbc[r * 128 + c]);
}

// ---------------------------------------------------------------------------
// dt = softplus(dt_raw + dt_b[d])  (in place)
// ---------------------------------------------------------------------------
__global__ void softplus_bias_kernel(float* __restrict__ dt,
                                     const float* __restrict__ dtb,
                                     int DI, size_t n) {
  size_t idx = (size_t)blockIdx.x * blockDim.x + threadIdx.x;
  if (idx >= n) return;
  int d = (int)(idx % DI);
  float v = dt[idx] + dtb[d];
  dt[idx] = (v > 20.f) ? v : log1pf(expf(v));
}

// ---------------------------------------------------------------------------
// Selective scan. One thread per (b, d); h[16] in registers, fp32.
// B/C rows (shared by all d of a batch) staged in LDS each step.
// rev=1 walks t from L-1 down to 0 (writes stay in physical order).
// dbc has ld 128: cols [64..79] = B, [80..95] = C.
// ---------------------------------------------------------------------------
__global__ __launch_bounds__(256) void scan_kernel(
    const float* __restrict__ dt, const float* __restrict__ u,
    const float* __restrict__ dbc, const float* __restrict__ Alog,
    float* __restrict__ y, int L, int DI, int rev) {
  int b = blockIdx.x;
  int d = blockIdx.y * blockDim.x + threadIdx.x;
  int tid = threadIdx.x;

  float Av[16];
#pragma unroll
  for (int nn = 0; nn < 16; nn++) Av[nn] = -expf(Alog[d * 16 + nn]);

  float h[16];
#pragma unroll
  for (int nn = 0; nn < 16; nn++) h[nn] = 0.f;

  __shared__ float sBC[32];   // [0..15] = B row, [16..31] = C row

  for (int tt = 0; tt < L; tt++) {
    int t = rev ? (L - 1 - tt) : tt;
    size_t bt = (size_t)b * L + t;
    __syncthreads();
    if (tid < 32) sBC[tid] = dbc[bt * 128 + 64 + tid];
    __syncthreads();
    float dtv = dt[bt * DI + d];
    float uv  = u[bt * DI + d];
    float du  = dtv * uv;
    float yv  = 0.f;
#pragma unroll
    for (int nn = 0; nn < 16; nn++) {
      float dA = expf(dtv * Av[nn]);
      h[nn] = dA * h[nn] + du * sBC[nn];
      yv += h[nn] * sBC[16 + nn];
    }
    y[bt * DI + d] = yv;
  }
}

// ---------------------------------------------------------------------------
// Gate: yz = bf16( (y + u*D[d]) * silu(z) ),  z = uz[..., DI + d]
// ---------------------------------------------------------------------------
__global__ void gate_kernel(const float* __restrict__ y,
                            const float* __restrict__ uc,
                            const float* __restrict__ uz,
                            const float* __restrict__ Dp,
                            __hip_bfloat16* __restrict__ yz_bf,
                            int DI, size_t n) {
  size_t idx = (size_t)blockIdx.x * blockDim.x + threadIdx.x;
  if (idx >= n) return;
  int d = (int)(idx % DI);
  size_t bt = idx / DI;
  float z = uz[bt * (size_t)(2 * DI) + DI + d];
  float val = (y[idx] + uc[idx] * Dp[d]) * (z / (1.f + expf(-z)));
  yz_bf[idx] = __float2bfloat16(val);
}

// ---------------------------------------------------------------------------
// out = x + out_f + out_b
// ---------------------------------------------------------------------------
__global__ void final_add_kernel(const float* __restrict__ x,
                                 const float* __restrict__ of,
                                 const float* __restrict__ ob,
                                 float* __restrict__ out, size_t n) {
  size_t i = (size_t)blockIdx.x * blockDim.x + threadIdx.x;
  if (i < n) out[i] = x[i] + of[i] + ob[i];
}

// ---------------------------------------------------------------------------
// Host orchestration
// ---------------------------------------------------------------------------
extern "C" void kernel_launch(void* const* d_in, const int* in_sizes, int n_in,
                              void* d_out, int out_size, void* d_ws, size_t ws_size,
                              hipStream_t stream) {
  (void)in_sizes; (void)n_in; (void)out_size; (void)ws_size;
  const int B = 2, L = 1024, DM = 1024, DI = 2048, DS = 16, DR = 64;
  const int M   = B * L;        // 2048 rows
  const int NIN = 2 * DI;       // 4096
  const int NX  = DR + 2 * DS;  // 96 real rows of xproj_w
  const int NXP = 128;          // padded to a multiple of 64 for the GEMM

  const float* x    = (const float*)d_in[0];
  const float* ln_g = (const float*)d_in[1];
  const float* ln_b = (const float*)d_in[2];
  // per-direction params: dir 0 = forward (indices 3..11), dir 1 = backward (12..20)
  const float* in_w[2]    = { (const float*)d_in[3],  (const float*)d_in[12] };
  const float* conv_w[2]  = { (const float*)d_in[4],  (const float*)d_in[13] };
  const float* conv_b[2]  = { (const float*)d_in[5],  (const float*)d_in[14] };
  const float* xproj_w[2] = { (const float*)d_in[6],  (const float*)d_in[15] };
  const float* dt_w[2]    = { (const float*)d_in[7],  (const float*)d_in[16] };
  const float* dt_b[2]    = { (const float*)d_in[8],  (const float*)d_in[17] };
  const float* Alog[2]    = { (const float*)d_in[9],  (const float*)d_in[18] };
  const float* Dp[2]      = { (const float*)d_in[10], (const float*)d_in[19] };
  const float* out_w[2]   = { (const float*)d_in[11], (const float*)d_in[20] };
  float* out = (float*)d_out;

  // ---- workspace bump allocator ----
  char* p = (char*)d_ws;
  auto alloc = [&](size_t bytes) -> char* {
    char* r = p;
    p += (bytes + 255) & ~(size_t)255;
    return r;
  };
  float*          xn     = (float*)         alloc((size_t)M * DM * 4);
  __hip_bfloat16* xn_bf  = (__hip_bfloat16*)alloc((size_t)M * DM * 2);
  __hip_bfloat16* w_in_bf[2], *w_xp_bf[2], *w_dt_bf[2], *w_out_bf[2];
  for (int d = 0; d < 2; d++) {
    w_in_bf[d]  = (__hip_bfloat16*)alloc((size_t)NIN * DM * 2);
    w_xp_bf[d]  = (__hip_bfloat16*)alloc((size_t)NXP * DI * 2);  // rows 96..127 zero
    w_dt_bf[d]  = (__hip_bfloat16*)alloc((size_t)DI * DR * 2);
    w_out_bf[d] = (__hip_bfloat16*)alloc((size_t)DM * DI * 2);
  }
  float*          uz      = (float*)         alloc((size_t)M * NIN * 4);  // reused per dir
  float*          uc      = (float*)         alloc((size_t)M * DI * 4);
  __hip_bfloat16* uc_bf   = (__hip_bfloat16*)alloc((size_t)M * DI * 2);
  float*          dbc     = (float*)         alloc((size_t)M * NXP * 4);  // ld 128
  __hip_bfloat16* dtin_bf = (__hip_bfloat16*)alloc((size_t)M * DR * 2);
  float*          dtbuf   = (float*)         alloc((size_t)M * DI * 4);
  float*          ybuf    = (float*)         alloc((size_t)M * DI * 4);
  __hip_bfloat16* yz_bf   = (__hip_bfloat16*)alloc((size_t)M * DI * 2);
  float*          outbuf[2];
  outbuf[0] = (float*)alloc((size_t)M * DM * 4);
  outbuf[1] = (float*)alloc((size_t)M * DM * 4);

  auto cvt = [&](const float* s, __hip_bfloat16* d, size_t n) {
    f32_to_bf16_kernel<<<dim3((unsigned)((n + 255) / 256)), dim3(256), 0, stream>>>(s, d, n);
  };
  auto gemm = [&](const __hip_bfloat16* A, const __hip_bfloat16* W, float* C,
                  int m, int n, int k) {
    dim3 grid(m / 16, (n + 511) / 512);
    gemm_bf16_nt_kernel<<<grid, dim3(256), 0, stream>>>(A, W, C, m, n, k);
  };

  // 1) weight conversion to bf16 (xproj zero-padded to 128 rows)
  for (int d = 0; d < 2; d++) {
    cvt(in_w[d],  w_in_bf[d],  (size_t)NIN * DM);
    cvt(dt_w[d],  w_dt_bf[d],  (size_t)DI * DR);
    cvt(out_w[d], w_out_bf[d], (size_t)DM * DI);
    size_t nxp = (size_t)NXP * DI;
    f32_to_bf16_pad_kernel<<<dim3((unsigned)((nxp + 255) / 256)), dim3(256), 0,
                             stream>>>(xproj_w[d], w_xp_bf[d], NX, DI, nxp);
  }

  // 2) layernorm -> xn (fp32 + bf16)
  layernorm_kernel<<<dim3(M), dim3(256), 0, stream>>>(x, ln_g, ln_b, xn, xn_bf, DM);

  // 3) both directions, serialized over shared scratch
  const size_t nMDI = (size_t)M * DI;
  for (int dir = 0; dir < 2; dir++) {
    // in-proj: uz[M, 4096] = xn @ in_w^T
    gemm(xn_bf, w_in_bf[dir], uz, M, NIN, DM);
    // depthwise conv + SiLU (direction-aware indexing)
    conv_silu_kernel<<<dim3((unsigned)((nMDI + 255) / 256)), dim3(256), 0, stream>>>(
        uz, conv_w[dir], conv_b[dir], uc, uc_bf, L, DI, dir, nMDI);
    // x-proj (padded): dbc[M, 128] = silu(conv(u)) @ xproj_w_pad^T
    gemm(uc_bf, w_xp_bf[dir], dbc, M, NXP, DI);
    // dt-proj input (first 64 cols of dbc) -> bf16
    extract_dtin_kernel<<<dim3((unsigned)(((size_t)M * DR + 255) / 256)), dim3(256),
                          0, stream>>>(dbc, dtin_bf, (size_t)M * DR);
    // dt_raw[M, 2048] = dtin @ dt_w^T ; then softplus(+bias) in place
    gemm(dtin_bf, w_dt_bf[dir], dtbuf, M, DI, DR);
    softplus_bias_kernel<<<dim3((unsigned)((nMDI + 255) / 256)), dim3(256), 0, stream>>>(
        dtbuf, dt_b[dir], DI, nMDI);
    // selective scan (sequential in t; reversed walk for dir==1)
    scan_kernel<<<dim3(B, DI / 256), dim3(256), 0, stream>>>(
        dtbuf, uc, dbc, Alog[dir], ybuf, L, DI, dir);
    // gate: (y + u*D) * silu(z) -> bf16
    gate_kernel<<<dim3((unsigned)((nMDI + 255) / 256)), dim3(256), 0, stream>>>(
        ybuf, uc, uz, Dp[dir], yz_bf, DI, nMDI);
    // out-proj: outbuf[M, 1024] = yz @ out_w^T
    gemm(yz_bf, w_out_bf[dir], outbuf[dir], M, DM, DI);
  }

  // 4) residual sum
  const size_t nOut = (size_t)M * DM;
  final_add_kernel<<<dim3((unsigned)((nOut + 255) / 256)), dim3(256), 0, stream>>>(
      x, outbuf[0], outbuf[1], out, nOut);
}
